// Spann3R_62388694941903
// MI455X (gfx1250) — matrix operand order repaired and checked by reading, not compiled, not run on hardware
//
#include <hip/hip_runtime.h>
#include <hip/hip_bf16.h>

#define B_ 4
#define P_ 1024
#define X_ 8192
#define C_ 1024
#define LN_EPS_ 1e-5f
#define ATTN_THRESH_ 0.0005f
#define INV_SQRT_C_ 0.03125f   // 1/sqrt(1024)

typedef float  v8f   __attribute__((ext_vector_type(8)));
typedef __bf16 v16bf __attribute__((ext_vector_type(16)));
typedef __bf16 v8bf  __attribute__((ext_vector_type(8)));

static __device__ __forceinline__ unsigned short f2bf_bits(float f) {
    union { __bf16 b; unsigned short u; } cv;
    cv.b = (__bf16)f;
    return cv.u;
}

static __device__ __forceinline__ float redmax16(float v) {
#pragma unroll
    for (int m = 8; m >= 1; m >>= 1) v = fmaxf(v, __shfl_xor(v, m, 16));
    return v;
}
static __device__ __forceinline__ float redsum16(float v) {
#pragma unroll
    for (int m = 8; m >= 1; m >>= 1) v += __shfl_xor(v, m, 16);
    return v;
}
static __device__ __forceinline__ float redsum32(float v) {
#pragma unroll
    for (int m = 16; m >= 1; m >>= 1) v += __shfl_xor(v, m, 32);
    return v;
}

// Async global->LDS copy of 16B (CDNA5 GVS mode: saddr base + per-lane voffset;
// VDST holds the LDS byte address). Tracked by ASYNCcnt.
static __device__ __forceinline__ void async_copy_b128(
    unsigned long long gbase, int goff_bytes, unsigned int lds_addr)
{
    asm volatile("global_load_async_to_lds_b128 %0, %1, %2 offset:0"
                 :
                 : "v"(lds_addr), "v"(goff_bytes), "s"(gbase)
                 : "memory");
}
static __device__ __forceinline__ void wait_asynccnt0() {
    asm volatile("s_wait_asynccnt 0x0" ::: "memory");
}

// ---------------------------------------------------------------------------
// Kernel 1: row layernorm f32 -> bf16, one wave32 per row of length C_=1024.
// block = 256 threads = 8 waves = 8 rows.
// ---------------------------------------------------------------------------
__global__ void __launch_bounds__(256) ln_rows_kernel(
    const float* __restrict__ in, const float* __restrict__ gam,
    const float* __restrict__ bet, unsigned short* __restrict__ out, int nrows)
{
    int row  = blockIdx.x * 8 + (threadIdx.x >> 5);
    int lane = threadIdx.x & 31;
    if (row >= nrows) return;

    const float* p = in + (size_t)row * C_;
    float4 xv[8];
    float s = 0.f, s2 = 0.f;
#pragma unroll
    for (int i = 0; i < 8; ++i) {
        int c = i * 128 + lane * 4;
        xv[i] = *(const float4*)(p + c);
        s  += xv[i].x + xv[i].y + xv[i].z + xv[i].w;
        s2 += xv[i].x * xv[i].x + xv[i].y * xv[i].y +
              xv[i].z * xv[i].z + xv[i].w * xv[i].w;
    }
    s  = redsum32(s);
    s2 = redsum32(s2);
    float mu  = s * (1.0f / C_);
    float var = s2 * (1.0f / C_) - mu * mu;
    float rs  = rsqrtf(var + LN_EPS_);

    unsigned short* o = out + (size_t)row * C_;
#pragma unroll
    for (int i = 0; i < 8; ++i) {
        int c = i * 128 + lane * 4;
        float4 gv = *(const float4*)(gam + c);
        float4 bv = *(const float4*)(bet + c);
        ushort4 o4;
        o4.x = f2bf_bits((xv[i].x - mu) * rs * gv.x + bv.x);
        o4.y = f2bf_bits((xv[i].y - mu) * rs * gv.y + bv.y);
        o4.z = f2bf_bits((xv[i].z - mu) * rs * gv.z + bv.z);
        o4.w = f2bf_bits((xv[i].w - mu) * rs * gv.w + bv.w);
        *(ushort4*)(o + c) = o4;
    }
}

// ---------------------------------------------------------------------------
// Kernel 2: layernorm V + transpose -> Vt[b][c][x] bf16.
// block = 256 threads handles 32 rows (x) of one batch; 32x32 LDS tiles.
// ---------------------------------------------------------------------------
__global__ void __launch_bounds__(256) ln_tr_v_kernel(
    const float* __restrict__ vin, const float* __restrict__ gam,
    const float* __restrict__ bet, unsigned short* __restrict__ vt)
{
    int x0g = blockIdx.x * 32;            // global row over [B*X]
    int b   = x0g >> 13;                  // / X_
    int x0  = x0g & (X_ - 1);
    int tid = threadIdx.x;
    int w   = tid >> 5, lane = tid & 31;

    __shared__ float mu_s[32], rs_s[32];
    __shared__ unsigned short tile_u[32 * 33];

    // phase 1: row stats (each wave does 4 rows)
#pragma unroll
    for (int rr = 0; rr < 4; ++rr) {
        int r = w * 4 + rr;
        const float* p = vin + (size_t)(x0g + r) * C_;
        float s = 0.f, s2 = 0.f;
#pragma unroll
        for (int i = 0; i < 8; ++i) {
            int c = i * 128 + lane * 4;
            float4 x4 = *(const float4*)(p + c);
            s  += x4.x + x4.y + x4.z + x4.w;
            s2 += x4.x * x4.x + x4.y * x4.y + x4.z * x4.z + x4.w * x4.w;
        }
        s  = redsum32(s);
        s2 = redsum32(s2);
        if (lane == 0) {
            float mu  = s * (1.0f / C_);
            float var = s2 * (1.0f / C_) - mu * mu;
            mu_s[r] = mu;
            rs_s[r] = rsqrtf(var + LN_EPS_);
        }
    }
    __syncthreads();

    // phase 2: 32x32 tiles, LN + transpose
    int tx = tid & 31, ty = tid >> 5;
    for (int ct = 0; ct < C_ / 32; ++ct) {
        int c = ct * 32 + tx;
        float gv = gam[c], bv = bet[c];
        float vals[4];
#pragma unroll
        for (int ry = 0; ry < 4; ++ry) {
            int r = ty + ry * 8;
            float x = vin[(size_t)(x0g + r) * C_ + c];
            vals[ry] = (x - mu_s[r]) * rs_s[r] * gv + bv;
        }
        __syncthreads();  // previous tile reads done
#pragma unroll
        for (int ry = 0; ry < 4; ++ry)
            tile_u[(ty + ry * 8) * 33 + tx] = f2bf_bits(vals[ry]);
        __syncthreads();
#pragma unroll
        for (int ry = 0; ry < 4; ++ry) {
            int cc = ct * 32 + ty + ry * 8;
            vt[(size_t)(b * C_ + cc) * X_ + x0 + tx] = tile_u[tx * 33 + ty + ry * 8];
        }
    }
}

// ---------------------------------------------------------------------------
// Kernel 3 (pass 1): S = (Qln . Kln^T) * invsqrtC * conf  via WMMA bf16,
// plus per-row softmax max M and normalizer Z (online across x).
// grid = B * P/16 blocks; block = 256 (8 waves, each owns X/8 = 1024 cols).
// Q tile is staged into LDS with async global->LDS loads (ASYNCcnt).
// ---------------------------------------------------------------------------
__global__ void __launch_bounds__(256, 1) attn_pass1(
    const unsigned short* __restrict__ qln, const unsigned short* __restrict__ kln,
    const float* __restrict__ conf, float* __restrict__ S,
    float* __restrict__ Mrow, float* __restrict__ Zrow)
{
    int blk = blockIdx.x;
    int b   = blk >> 6;
    int p0  = (blk & 63) << 4;

    __shared__ unsigned short qs_u[16 * 1024];       // 32KB Q tile (bf16)
    __shared__ float m_s[8][16], z_s[8][16];

    int tid = threadIdx.x;
    // stage Q tile: 16 rows are contiguous in Qln -> async 32KB global->LDS copy
    {
        unsigned long long gbase =
            (unsigned long long)(uintptr_t)(qln + (size_t)(b * P_ + p0) * C_);
        unsigned int lds_base = (unsigned int)(uintptr_t)qs_u;
#pragma unroll
        for (int k = 0; k < 8; ++k) {
            int off = (tid + k * 256) * 16;
            async_copy_b128(gbase, off, lds_base + off);
        }
        wait_asynccnt0();
    }
    __syncthreads();

    int w = tid >> 5, lane = tid & 31;
    int h = lane >> 4, n = lane & 15, m = lane & 15;

    const __bf16* qs = (const __bf16*)qs_u;
    const __bf16* K  = (const __bf16*)kln;

    float m8[8], z8[8];
#pragma unroll
    for (int r = 0; r < 8; ++r) { m8[r] = -1e30f; z8[r] = 0.f; }

    const __bf16* qrow = qs + m * 1024 + 16 * h;

    for (int xt = 0; xt < 64; ++xt) {
        int x0w = w * 1024 + xt * 16;
        float cf = conf[(size_t)b * X_ + x0w + n];

        const __bf16* krow = K + (size_t)(b * X_ + x0w + n) * C_ + 16 * h;
        __builtin_prefetch((const void*)(krow + 16 * C_), 0, 1);

        v8f acc = {};
#pragma unroll 8
        for (int kc = 0; kc < 32; ++kc) {
            union { v16bf v; v8bf g8[2]; } A;
            A.g8[0] = *(const v8bf*)(qrow + kc * 32);
            A.g8[1] = *(const v8bf*)(qrow + kc * 32 + 8);
            v16bf Bv = *(const v16bf*)(krow + kc * 32);
            acc = __builtin_amdgcn_wmma_f32_16x16x32_bf16(
                false, A.v, false, Bv, (short)0, acc, false, false);
        }

        size_t srow = (size_t)(b * P_ + p0 + 8 * h) * X_ + x0w + n;
#pragma unroll
        for (int r = 0; r < 8; ++r) {
            float s = acc[r] * INV_SQRT_C_ * cf;
            S[srow + (size_t)r * X_] = s;
            float tmax = redmax16(s);
            float tsum = redsum16(__expf(s - tmax));
            float nm = fmaxf(m8[r], tmax);
            z8[r] = z8[r] * __expf(m8[r] - nm) + tsum * __expf(tmax - nm);
            m8[r] = nm;
        }
    }

    if (n == 0) {
#pragma unroll
        for (int r = 0; r < 8; ++r) {
            m_s[w][8 * h + r] = m8[r];
            z_s[w][8 * h + r] = z8[r];
        }
    }
    __syncthreads();
    if (tid < 16) {
        float mm = -1e30f;
#pragma unroll
        for (int w2 = 0; w2 < 8; ++w2) mm = fmaxf(mm, m_s[w2][tid]);
        float zz = 0.f;
#pragma unroll
        for (int w2 = 0; w2 < 8; ++w2) zz += z_s[w2][tid] * __expf(m_s[w2][tid] - mm);
        Mrow[b * P_ + p0 + tid] = mm;
        Zrow[b * P_ + p0 + tid] = zz;
    }
}

// ---------------------------------------------------------------------------
// Kernel 4 (pass 2): p = exp(S - M); zero if p < thresh*Z; out = (p.V)/sum(p)
// + feat. grid = B * P/16; block = 256: waves build a shared 16x128 prob tile
// (double-buffered in LDS), then each wave owns 128 output channels for the
// attn.V WMMAs against Vt.
// ---------------------------------------------------------------------------
__global__ void __launch_bounds__(256, 1) attn_pass2(
    const float* __restrict__ S, const unsigned short* __restrict__ vt,
    const float* __restrict__ feat, const float* __restrict__ Mrow,
    const float* __restrict__ Zrow, float* __restrict__ out)
{
    int blk = blockIdx.x;
    int b   = blk >> 6;
    int p0  = (blk & 63) << 4;

    __shared__ unsigned short at_u[2][16 * 128];     // prob tiles, bf16
    __shared__ float den_s[8][16];
    __shared__ float denf[16];

    int tid = threadIdx.x, w = tid >> 5, lane = tid & 31;
    int h = lane >> 4, n = lane & 15, m = lane & 15;
    int cbase = w * 128;

    float m_r[8], th_r[8], den8[8];
#pragma unroll
    for (int r = 0; r < 8; ++r) {
        int row = b * P_ + p0 + 8 * h + r;
        m_r[r]  = Mrow[row];
        th_r[r] = ATTN_THRESH_ * Zrow[row];
        den8[r] = 0.f;
    }

    v8f acc[8];
#pragma unroll
    for (int ct = 0; ct < 8; ++ct) acc[ct] = (v8f){};

    const __bf16* V = (const __bf16*)vt;

    for (int it = 0; it < 64; ++it) {
        int xmac = it * 128;
        int buf  = it & 1;

        // stage 1: this wave's 16-column slice of the 16x128 prob tile
        {
            int xs = xmac + w * 16;
            size_t srow = (size_t)(b * P_ + p0 + 8 * h) * X_ + xs + n;
            __bf16* atb = (__bf16*)at_u[buf];
#pragma unroll
            for (int r = 0; r < 8; ++r) {
                float s = S[srow + (size_t)r * X_];
                float p = __expf(s - m_r[r]);
                p = (p >= th_r[r]) ? p : 0.f;
                den8[r] += p;
                atb[(8 * h + r) * 128 + w * 16 + n] = (__bf16)p;
            }
        }
        __syncthreads();

        // stage 2: attn-tile x Vt  for this wave's 128 channels
        {
            const __bf16* atb = (const __bf16*)at_u[buf];
#pragma unroll
            for (int xt = 0; xt < 4; ++xt) {
                union { v16bf v; v8bf g8[2]; } A;
                const __bf16* ar = atb + m * 128 + xt * 32 + 16 * h;
                A.g8[0] = *(const v8bf*)ar;
                A.g8[1] = *(const v8bf*)(ar + 8);
                size_t vbase = (size_t)(b * C_ + cbase + n) * X_ + xmac + xt * 32 + 16 * h;
#pragma unroll
                for (int ct = 0; ct < 8; ++ct) {
                    v16bf Bv = *(const v16bf*)(V + vbase + (size_t)ct * 16 * X_);
                    acc[ct] = __builtin_amdgcn_wmma_f32_16x16x32_bf16(
                        false, A.v, false, Bv, (short)0, acc[ct], false, false);
                }
            }
        }
    }

    // reduce thresholded row sums across lanes and waves
#pragma unroll
    for (int r = 0; r < 8; ++r) {
        float d = redsum16(den8[r]);
        if (n == 0) den_s[w][8 * h + r] = d;
    }
    __syncthreads();
    if (tid < 16) {
        float s = 0.f;
#pragma unroll
        for (int w2 = 0; w2 < 8; ++w2) s += den_s[w2][tid];
        denf[tid] = s;
    }
    __syncthreads();

    float ir[8];
#pragma unroll
    for (int r = 0; r < 8; ++r) ir[r] = 1.0f / denf[8 * h + r];

#pragma unroll
    for (int ct = 0; ct < 8; ++ct) {
#pragma unroll
        for (int r = 0; r < 8; ++r) {
            size_t idx = (size_t)(b * P_ + p0 + 8 * h + r) * C_ + cbase + ct * 16 + n;
            out[idx] = acc[ct][r] * ir[r] + feat[idx];
        }
    }
}

// ---------------------------------------------------------------------------
extern "C" void kernel_launch(void* const* d_in, const int* in_sizes, int n_in,
                              void* d_out, int out_size, void* d_ws, size_t ws_size,
                              hipStream_t stream)
{
    const float* feat  = (const float*)d_in[0];
    const float* mem_k = (const float*)d_in[1];
    const float* mem_v = (const float*)d_in[2];
    const float* mem_c = (const float*)d_in[3];
    // d_in[4] (mem_attn) is unused by the reference output
    const float* g_q = (const float*)d_in[5];
    const float* b_q = (const float*)d_in[6];
    const float* g_k = (const float*)d_in[7];
    const float* b_k = (const float*)d_in[8];
    const float* g_v = (const float*)d_in[9];
    const float* b_v = (const float*)d_in[10];
    float* out = (float*)d_out;

    char* w = (char*)d_ws;
    unsigned short* Qln = (unsigned short*)w; w += (size_t)B_ * P_ * C_ * 2;   //   8 MB
    unsigned short* Kln = (unsigned short*)w; w += (size_t)B_ * X_ * C_ * 2;   //  64 MB
    unsigned short* Vt  = (unsigned short*)w; w += (size_t)B_ * C_ * X_ * 2;   //  64 MB
    float* Smat = (float*)w;                  w += (size_t)B_ * P_ * X_ * 4;   // 128 MB
    float* Mrow = (float*)w;                  w += (size_t)B_ * P_ * 4;
    float* Zrow = (float*)w;

    ln_rows_kernel<<<(B_ * P_) / 8, 256, 0, stream>>>(feat, g_q, b_q, Qln, B_ * P_);
    ln_rows_kernel<<<(B_ * X_) / 8, 256, 0, stream>>>(mem_k, g_k, b_k, Kln, B_ * X_);
    ln_tr_v_kernel<<<(B_ * X_) / 32, 256, 0, stream>>>(mem_v, g_v, b_v, Vt);
    attn_pass1<<<(B_ * P_) / 16, 256, 0, stream>>>(Qln, Kln, mem_c, Smat, Mrow, Zrow);
    attn_pass2<<<(B_ * P_) / 16, 256, 0, stream>>>(Smat, Vt, feat, Mrow, Zrow, out);
}
